// tfphasor_atten_fusion_20177756356736
// MI455X (gfx1250) — matrix operand order repaired and checked by reading, not compiled, not run on hardware
//
#include <hip/hip_runtime.h>
#include <hip/hip_bf16.h>
#include <math.h>

// ---------------------------------------------------------------------------
// tfphasor_atten_fusion for MI455X (gfx1250, wave32)
//   feature (2,1,128,64,64) f32 -> out (2,1,128,64,64) f32
// Pipeline: wave conv1d -> WMMA GEMM (mtx) -> pad -> 4D FFT -> freq-atten
//           multiply -> 4D iFFT -> crop/combine -> WMMA GEMM (mtxi)
// GEMM uses async global->LDS staging (ASYNCcnt) + v_wmma_f32_16x16x32_f16.
// ---------------------------------------------------------------------------

typedef __attribute__((ext_vector_type(16))) _Float16 v16h;
typedef __attribute__((ext_vector_type(8)))  float    v8f;

#define BD      2          // b*d
#define T_DIM   128
#define H_DIM   64
#define W_DIM   64
#define HW      4096       // H*W
#define PT      256        // padded T
#define PH      128        // padded H
#define PW      128        // padded W
#define VOL_P   4194304    // PT*PH*PW (one padded batch volume, complex elems)
#define NB      4          // 2*BD batches in padded volume
#define KTAPS   63
#define ATT_N   524288     // 128*64*64 attention region size
#define PI_F    3.14159265358979323846f

__device__ __forceinline__ float2 cadd(float2 a, float2 b){ return make_float2(a.x+b.x, a.y+b.y); }
__device__ __forceinline__ float2 csub(float2 a, float2 b){ return make_float2(a.x-b.x, a.y-b.y); }
__device__ __forceinline__ float2 cmul(float2 a, float2 b){
  return make_float2(a.x*b.x - a.y*b.y, a.x*b.y + a.y*b.x);
}

// ---------------------------------------------------------------------------
// 1) 63-tap conv along T (2 channels: cos/sin).  One block per (bd,h,w) line.
//    data2 layout: (4, T, HW)  with channel c at batch index c*BD+bd.
// ---------------------------------------------------------------------------
__global__ void conv_wave_kernel(const float* __restrict__ feat,
                                 const float* __restrict__ wave,
                                 float* __restrict__ data2) {
  __shared__ float xs[T_DIM];
  __shared__ float wk[2*KTAPS];
  const int pix = blockIdx.x;          // bd*HW + hw
  const int bd  = pix >> 12;
  const int hw  = pix & (HW - 1);
  const int t   = threadIdx.x;

  xs[t] = feat[(long)bd * T_DIM * HW + (long)t * HW + hw];
  for (int i = t; i < 2*KTAPS; i += blockDim.x) wk[i] = wave[i];
  __syncthreads();

  float acc0 = 0.f, acc1 = 0.f;
  for (int i = 0; i < KTAPS; ++i) {
    const int src = t + i - (KTAPS/2);
    const float xv = (src >= 0 && src < T_DIM) ? xs[src] : 0.f;
    acc0 = fmaf(xv, wk[i], acc0);
    acc1 = fmaf(xv, wk[KTAPS + i], acc1);
  }
  data2[((long)(0*BD + bd) * T_DIM + t) * HW + hw] = acc0;
  data2[((long)(1*BD + bd) * T_DIM + t) * HW + hw] = acc1;
}

// ---------------------------------------------------------------------------
// 2/9) WMMA GEMM:  C[b,t,x] = sum_m A[t,m] * B[b,m,x]
//   Block: 256 threads = 8 waves; tile = 16 t-rows x 128 x-cols, K = 128.
//   A tile (16x128 f32) staged once to LDS (row padded to 132 floats).
//   B slabs (32x128 f32, 16 KB) streamed with global_load_async_to_lds_b128
//   (ASYNCcnt), double-buffered; fragments built from LDS with f32->f16 cvt.
//   Fragment layouts per CDNA5 ISA 7.12.2 (wave32).
// ---------------------------------------------------------------------------
#define APAD 132
__global__ void wmma_gemm_kernel(const float* __restrict__ A,
                                 const float* __restrict__ Bm,
                                 float* __restrict__ C,
                                 int xcols) {
  __shared__ float As[16 * APAD];     // ~8.4 KB
  __shared__ float Bs[2][32 * 128];   // 32 KB (double buffer)

  const int tid   = threadIdx.x;
  const int lane  = tid & 31;
  const int wave  = tid >> 5;
  const int x0b   = blockIdx.x * 128;
  const int t0    = blockIdx.y * 16;
  const int b     = blockIdx.z;
  const int xl    = lane & 15;        // col within wave's 16-wide tile / A row M
  const int khalf = lane >> 4;
  const int xw    = wave * 16;        // wave's col offset within block tile

  const float* __restrict__ Bb = Bm + (long)b * 128 * xcols;
  float*       __restrict__ Cb = C  + (long)b * 128 * xcols;

  // ---- stage A tile: 16 rows x 128 K, coalesced b128 loads --------------
  {
    const int r = tid >> 4;           // 0..15
    const int c = (tid & 15) * 8;     // 0,8,...,120
    const float4* src = (const float4*)&A[(long)(t0 + r) * 128 + c];
    float4 v0 = src[0], v1 = src[1];
    *(float4*)&As[r * APAD + c]     = v0;
    *(float4*)&As[r * APAD + c + 4] = v1;
  }

  // ---- async stage of one 32x128 f32 B slab into Bs[buf] ----------------
  auto stage = [&](int slab, int buf) {
    const unsigned bsb = (unsigned)(unsigned long long)(const void*)&Bs[buf][0];
    for (int c = tid; c < 1024; c += 256) {       // 1024 chunks of 16 B
      const int row = c >> 5;                     // 0..31 (K within slab)
      const int ch  = c & 31;                     // 16B chunk within row
      const float* g = &Bb[(long)(slab * 32 + row) * xcols + x0b + ch * 4];
      const unsigned lds = bsb + (unsigned)((row * 128 + ch * 4) * 4);
      asm volatile("global_load_async_to_lds_b128 %0, %1, off"
                   :: "v"(lds), "v"((unsigned long long)(const void*)g)
                   : "memory");
    }
  };

  v8f acc = {};
  stage(0, 0);
#pragma unroll
  for (int s = 0; s < 4; ++s) {
    asm volatile("s_wait_asynccnt 0x0" ::: "memory");  // this wave's stages done
    __syncthreads();                                   // all waves' stages done
    if (s < 3) stage(s + 1, (s + 1) & 1);              // prefetch next slab

    const float* __restrict__ Bsl = Bs[s & 1];
    const int k0 = s * 32;

    v16h a, bf;
#pragma unroll
    for (int j = 0; j < 8; ++j) {
      // A fragment: lane row M = xl; K pairs per ISA 16-bit A layout
      const int kb = k0 + ((j & 4) ? 16 : 0) + 2 * (j & 3) + 8 * khalf;
      a[2*j]   = (_Float16)As[xl * APAD + kb];
      a[2*j+1] = (_Float16)As[xl * APAD + kb + 1];
    }
#pragma unroll
    for (int i = 0; i < 16; ++i) {
      // B fragment: K rows in same order as A (8*khalf + i, +16 for i>=8)
      bf[i] = (_Float16)Bsl[((i & 7) + 8 * khalf + 16 * (i >> 3)) * 128 + xw + xl];
    }
    acc = __builtin_amdgcn_wmma_f32_16x16x32_f16(
        /*neg_a=*/false, a, /*neg_b=*/false, bf,
        /*c_mod=*/(short)0, acc, /*reuse_a=*/false, /*reuse_b=*/false);
  }

#pragma unroll
  for (int r = 0; r < 8; ++r)
    Cb[(long)(t0 + r + 8 * khalf) * xcols + (x0b + xw + xl)] = acc[r];
}

// ---------------------------------------------------------------------------
// 3) zero-pad (4,128,64,64) -> complex (4,256,128,128)
// ---------------------------------------------------------------------------
__global__ void pad_kernel(const float* __restrict__ tmp, float2* __restrict__ F) {
  const long idx = (long)blockIdx.x * blockDim.x + threadIdx.x;  // < NB*VOL_P
  const int w = (int)(idx & 127);
  const int h = (int)((idx >> 7) & 127);
  const int t = (int)((idx >> 14) & 255);
  const int b = (int)(idx >> 22);
  float v = 0.f;
  if (t < T_DIM && h < H_DIM && w < W_DIM)
    v = tmp[((long)b * T_DIM + t) * HW + h * W_DIM + w];
  F[idx] = make_float2(v, 0.f);
}

// ---------------------------------------------------------------------------
// 4) frequency-attention scalar reductions (single block -> deterministic).
// Rolls in _fre_atten2 cancel, so only 6 scalars/channel are needed:
//   sc[0]=scal_l  sc[1]=M1  sc[2]=E1  sc[3]=scal_h  sc[4]=M2  sc[5]=E2
// ---------------------------------------------------------------------------
__global__ void atten_reduce_kernel(const float* __restrict__ psf,
                                    const float* __restrict__ pw,
                                    const float* __restrict__ pb,
                                    float* __restrict__ sc, int phase) {
  const float w = pw[0], b = pb[0];
  const bool is_max = (phase == 1) || (phase == 4);
  float acc = is_max ? -3.4e38f : 0.f;

  for (int a = threadIdx.x; a < ATT_N; a += blockDim.x) {
    const int aw = a & 63, ah = (a >> 6) & 63, at = a >> 12;
    long src;
    if (phase <= 3) {  // "low" = anti-center samples of psf
      const int st = (at + 192) & 255, sh = (ah + 96) & 127, sw = (aw + 96) & 127;
      src = ((long)st * 128 + sh) * 128 + sw;
    } else {           // "high" = center region
      src = ((long)(64 + at) * 128 + (32 + ah)) * 128 + (32 + aw);
    }
    const float kv  = psf[src];
    const float k2p = kv + __sinf((float)at);
    switch (phase) {
      case 0: acc += kv; break;
      case 1: acc = fmaxf(acc, sc[0] * (w * k2p + b)); break;
      case 2: acc += __expf(sc[0] * (w * k2p + b) - sc[1]); break;
      case 3: {
        const float sm = __expf(sc[0] * (w * k2p + b) - sc[1]) / sc[2];
        acc += (1.f / (1.f + __expf(-sm))) * k2p;    // sum of l2l
      } break;
      case 4: acc = fmaxf(acc, sc[3] * (w * k2p + b)); break;
      case 5: acc += __expf(sc[3] * (w * k2p + b) - sc[4]); break;
    }
  }
  __shared__ float red[256];
  red[threadIdx.x] = acc;
  __syncthreads();
  for (int s = 128; s > 0; s >>= 1) {
    if ((int)threadIdx.x < s)
      red[threadIdx.x] = is_max ? fmaxf(red[threadIdx.x], red[threadIdx.x + s])
                                : red[threadIdx.x] + red[threadIdx.x + s];
    __syncthreads();
  }
  if (threadIdx.x == 0) {
    const float r = red[0];
    if (phase == 0)      sc[0] = w * (r / (float)ATT_N) + b;   // scal_l
    else if (phase == 3) sc[3] = w * (r / (float)ATT_N) + b;   // scal_h
    else                 sc[phase] = r;
  }
}

// ---------------------------------------------------------------------------
// 5/7) in-place Stockham radix-2 FFT over one axis; stride == inner size.
//      One line per N/2-thread group (LINES lines per block), ping-pong in LDS.
// ---------------------------------------------------------------------------
template <int N, int LINES>
__global__ void fft_lines_kernel(float2* __restrict__ F, int stride, int inverse) {
  __shared__ float2 bufA[LINES][N];
  __shared__ float2 bufB[LINES][N];
  const int half = N / 2;
  const int lb = threadIdx.x / half;
  const int j  = threadIdx.x % half;
  const long line = (long)blockIdx.x * LINES + lb;
  const long base = (line / stride) * (long)N * stride + (line % stride);

  float2* x = bufA[lb];
  float2* y = bufB[lb];
  x[j]        = F[base + (long)j * stride];
  x[j + half] = F[base + (long)(j + half) * stride];
  __syncthreads();

  const float sgn = inverse ? 1.f : -1.f;
  for (int Ns = 1; Ns < N; Ns *= 2) {
    const int  r     = j & (Ns - 1);
    const float ang  = sgn * 2.f * PI_F * (float)r / (float)(2 * Ns);
    float2 tw;  tw.x = __cosf(ang);  tw.y = __sinf(ang);
    const float2 a = x[j];
    const float2 c = cmul(tw, x[j + half]);
    const int idxD = ((j / Ns) * Ns * 2) + r;
    y[idxD]      = cadd(a, c);
    y[idxD + Ns] = csub(a, c);
    __syncthreads();
    float2* t = x; x = y; y = t;
  }
  F[base + (long)j * stride]          = x[j];
  F[base + (long)(j + half) * stride] = x[j + half];
}

// length-4 DFT over the batch axis (elementwise radix-4)
__global__ void fft_batch4_kernel(float2* __restrict__ F, int inverse) {
  const long i = (long)blockIdx.x * blockDim.x + threadIdx.x;  // < VOL_P
  const long S = VOL_P;
  const float2 a = F[i], b = F[i + S], c = F[i + 2*S], d = F[i + 3*S];
  const float2 apc = cadd(a, c), amc = csub(a, c);
  const float2 bpd = cadd(b, d), bmd = csub(b, d);
  const float2 ib = inverse ? make_float2(-bmd.y,  bmd.x)    // +i*(b-d)
                            : make_float2( bmd.y, -bmd.x);   // -i*(b-d)
  F[i]       = cadd(apc, bpd);
  F[i + S]   = cadd(amc, ib);
  F[i + 2*S] = csub(apc, bpd);
  F[i + 3*S] = csub(amc, ib);
}

// ---------------------------------------------------------------------------
// 6) apply (w1 + i*w2) computed on the fly from invpsf + attention scalars.
// ---------------------------------------------------------------------------
__device__ __forceinline__ float atten_apply(float kv, int pe_idx, float w, float b,
                                             float scal, float mx, float se) {
  const float k2p  = kv + __sinf((float)pe_idx);
  const float sm   = __expf(scal * (w * k2p + b) - mx) / se;
  return (1.f / (1.f + __expf(-sm))) * k2p;
}

__global__ void freq_mul_kernel(float2* __restrict__ F,
                                const float* __restrict__ psf_r,
                                const float* __restrict__ psf_i,
                                const float* __restrict__ pw,
                                const float* __restrict__ pb,
                                const float* __restrict__ sc) {
  const long idx = (long)blockIdx.x * blockDim.x + threadIdx.x;  // < VOL_P
  const int wv = (int)(idx & 127);
  const int h  = (int)((idx >> 7) & 127);
  const int t  = (int)(idx >> 14);
  const bool tc = (t >= 64 && t < 192), hc = (h >= 32 && h < 96), wc = (wv >= 32 && wv < 96);

  const float cw = pw[0], cb = pb[0];
  const float kr = psf_r[idx], ki = psf_i[idx];
  float w1, w2;
  if (tc && hc && wc) {                       // center -> l2h
    const int pi = t - 64;
    w1 = atten_apply(kr, pi, cw, cb, sc[3],  sc[4],  sc[5]);
    w2 = atten_apply(ki, pi, cw, cb, sc[11], sc[12], sc[13]);
  } else if (!tc && !hc && !wc) {             // anti-center -> l2l
    const int pi = (t >= 192) ? (t - 192) : (t + 64);
    w1 = atten_apply(kr, pi, cw, cb, sc[0],  sc[1],  sc[2]);
    w2 = atten_apply(ki, pi, cw, cb, sc[8],  sc[9],  sc[10]);
  } else {                                    // untouched
    w1 = kr; w2 = ki;
  }
#pragma unroll
  for (int b = 0; b < NB; ++b) {
    const float2 v = F[(long)b * VOL_P + idx];
    F[(long)b * VOL_P + idx] =
        make_float2(v.x * w1 - v.y * w2, v.x * w2 + v.y * w1);
  }
}

// ---------------------------------------------------------------------------
// 8) crop to (:,128,64,64), ifft scale, magnitude combine -> sq (2,T,HW)
// ---------------------------------------------------------------------------
__global__ void combine_kernel(const float2* __restrict__ F, float* __restrict__ sq) {
  const long idx = (long)blockIdx.x * blockDim.x + threadIdx.x;  // < 2*T*HW
  const int wv = (int)(idx & 63);
  const int h  = (int)((idx >> 6) & 63);
  const int t  = (int)((idx >> 12) & 127);
  const int bd = (int)(idx >> 19);
  const long fi = ((long)t * 128 + h) * 128 + wv;
  const float scale = 1.f / 16777216.f;       // ifftn over 4*256*128*128

  const float2 cv = F[(long)bd * VOL_P + fi];          // cos channel
  const float2 sv = F[(long)(bd + BD) * VOL_P + fi];   // sin channel
  const float cr = cv.x * scale, ci = cv.y * scale;
  const float sr = sv.x * scale, si = sv.y * scale;
  const float sum_real = cr*cr - ci*ci + sr*sr - si*si;
  const float sum_imag = 2.f*cr*ci + 2.f*sr*si;
  const float mag = fmaxf(0.f, (sqrtf(sum_real*sum_real + sum_imag*sum_imag) + sum_real) * 0.5f);
  sq[idx] = sqrtf(mag);
}

// ---------------------------------------------------------------------------
// launcher
// ---------------------------------------------------------------------------
extern "C" void kernel_launch(void* const* d_in, const int* in_sizes, int n_in,
                              void* d_out, int out_size, void* d_ws, size_t ws_size,
                              hipStream_t stream) {
  (void)in_sizes; (void)n_in; (void)out_size; (void)ws_size;
  const float* feature = (const float*)d_in[0];
  const float* conv_w  = (const float*)d_in[1];
  const float* conv_b  = (const float*)d_in[2];
  const float* wave    = (const float*)d_in[3];
  const float* psf_r   = (const float*)d_in[4];
  const float* psf_i   = (const float*)d_in[5];
  const float* mtx     = (const float*)d_in[6];
  const float* mtxi    = (const float*)d_in[7];
  float* out = (float*)d_out;

  char* ws = (char*)d_ws;
  float2* F     = (float2*)(ws);                       // 134,217,728 B
  float*  data2 = (float*)(ws + 134217728);            //   8,388,608 B
  float*  tmp   = (float*)(ws + 142606336);            //   8,388,608 B
  float*  sq    = (float*)(ws + 150994944);            //   4,194,304 B
  float*  sc    = (float*)(ws + 155189248);            //          64 B

  // 1) wave conv along T
  conv_wave_kernel<<<BD * HW, T_DIM, 0, stream>>>(feature, wave, data2);

  // 2) tmp[b,t,x] = mtx @ data2   (WMMA, async-LDS staged)
  wmma_gemm_kernel<<<dim3(HW / 128, T_DIM / 16, NB), 256, 0, stream>>>(mtx, data2, tmp, HW);

  // 3) zero-pad into complex volume
  pad_kernel<<<(NB * VOL_P) / 256, 256, 0, stream>>>(tmp, F);

  // 4) attention scalars (real then imag channel), sequential phases
  for (int ch = 0; ch < 2; ++ch) {
    const float* psf = ch ? psf_i : psf_r;
    for (int phase = 0; phase < 6; ++phase)
      atten_reduce_kernel<<<1, 256, 0, stream>>>(psf, conv_w, conv_b, sc + ch * 8, phase);
  }

  // 5) forward 4D FFT: W, H, T axes then batch-4 axis
  fft_lines_kernel<PW, 4><<<(NB * PT * PH) / 4, 256, 0, stream>>>(F, 1, 0);
  fft_lines_kernel<PH, 4><<<(NB * PT * PW) / 4, 256, 0, stream>>>(F, PW, 0);
  fft_lines_kernel<PT, 2><<<(NB * PH * PW) / 2, 256, 0, stream>>>(F, PH * PW, 0);
  fft_batch4_kernel<<<VOL_P / 256, 256, 0, stream>>>(F, 0);

  // 6) frequency-domain attention multiply
  freq_mul_kernel<<<VOL_P / 256, 256, 0, stream>>>(F, psf_r, psf_i, conv_w, conv_b, sc);

  // 7) inverse 4D FFT
  fft_batch4_kernel<<<VOL_P / 256, 256, 0, stream>>>(F, 1);
  fft_lines_kernel<PT, 2><<<(NB * PH * PW) / 2, 256, 0, stream>>>(F, PH * PW, 1);
  fft_lines_kernel<PH, 4><<<(NB * PT * PW) / 4, 256, 0, stream>>>(F, PW, 1);
  fft_lines_kernel<PW, 4><<<(NB * PT * PH) / 4, 256, 0, stream>>>(F, 1, 1);

  // 8) crop + magnitude combine
  combine_kernel<<<(BD * T_DIM * HW) / 256, 256, 0, stream>>>(F, sq);

  // 9) out[b,t,x] = mtxi @ sq   (WMMA, async-LDS staged)
  wmma_gemm_kernel<<<dim3(HW / 128, T_DIM / 16, BD), 256, 0, stream>>>(mtxi, sq, out, HW);
}
